// RegressorESN_80788334838318
// MI455X (gfx1250) — matrix operand order repaired and checked by reading, not compiled, not run on hardware
//
#include <hip/hip_runtime.h>
#include <cmath>

// ---------------------------------------------------------------------------
// ESN recurrence on MI455X (gfx1250):
//   h_t = tanh(U_t + W_h @ h_{t-1}),  out_t = dot(h_t, w_ro) + b_ro
// W_h kept bf16 + L2-resident (32MB << 192MB L2); matvec done with
// v_wmma_f32_16x16x32_bf16 (f32 accumulation), h replicated across B columns.
// This revision: software-pipelined inner loop (register double-buffering of
// A/B fragments -> partial LOADcnt waits instead of full drains), dual
// accumulators to break the WMMA D->C serial chain, branchless wrapped
// prefetch (no per-iteration s_cbranch).
// ---------------------------------------------------------------------------

typedef __attribute__((ext_vector_type(16))) __bf16 v16bf;
typedef __attribute__((ext_vector_type(8)))  float  v8f;

constexpr int H   = 4096;   // hidden
constexpr int L   = 4096;   // seq len
constexpr int IN  = 128;    // input size
constexpr int NWG = 32;     // persistent workgroups (32 x 8 waves = 256 waves)
constexpr int TPB = 256;    // threads per block (8 wave32)
constexpr int KT  = H / 32; // 128 K-tiles per row-tile
// (H/16 = 256 row tiles == 256 waves, one tile per wave)

// ---- workspace layout (bytes), all 256B aligned ---------------------------
constexpr size_t WS_COUNTER = 0;                                   // 1 x u32
constexpr size_t WS_H       = 256;                                 // 2*H bf16 (double buffer)
constexpr size_t WS_PART    = WS_H + (size_t)2 * H * 2;            // L*NWG f32 partial readouts
constexpr size_t WS_U       = WS_PART + (size_t)L * NWG * 4;       // L*H f32 (input projection)
constexpr size_t WS_WH      = WS_U + (size_t)L * H * 4;            // H*H bf16 (swizzled A frags)
// total ~101.2 MB

// ---------------------------------------------------------------------------
// Setup: zero barrier counter, init h double-buffer from h_init (bf16).
// ---------------------------------------------------------------------------
__global__ void esn_setup(const float* __restrict__ h_init, unsigned char* ws) {
  int i = blockIdx.x * blockDim.x + threadIdx.x;
  if (i == 0) *(unsigned int*)(ws + WS_COUNTER) = 0u;
  __bf16* hb = (__bf16*)(ws + WS_H);
  if (i < H) {
    hb[i]     = (__bf16)h_init[i];
    hb[H + i] = (__bf16)0.0f;
  }
}

// ---------------------------------------------------------------------------
// One-time: W_h (f32, row-major HxH, y[j] = sum_k Wh[j][k] h[k]) -> bf16,
// pre-swizzled into per-lane WMMA A-fragment order (16-bit A 16x32, contiguous k per lane half; B uses the same):
//   lanes 0-15 : row M=lane,     K in {0..15} of the 32-K tile
//   lanes 16-31: row M=lane-16,  K in {16..31}
// Fragment index: ((mt*KT + kt)*32 + lane)*16 + e   (16 bf16 = 32B per lane)
// ---------------------------------------------------------------------------
__global__ void esn_swizzle_wh(const float* __restrict__ Wh, unsigned char* ws) {
  size_t t = (size_t)blockIdx.x * blockDim.x + threadIdx.x;   // H*H threads
  int e    = (int)(t & 15);
  int lane = (int)((t >> 4) & 31);
  int kt   = (int)((t >> 9) & (KT - 1));
  int mt   = (int)(t >> 16);
  int half = lane >> 4;
  int j    = mt * 16 + (lane & 15);
  int kloc = e;                              // half 0: {0..15}
  int k    = kt * 32 + kloc + half * 16;     // half 1: {16..31}
  __bf16* dst = (__bf16*)(ws + WS_WH);
  dst[t] = (__bf16)Wh[(size_t)j * H + k];
}

// ---------------------------------------------------------------------------
// One-time: U[t][j] = bias[j] + sum_k input[t][k] * W_in[j][k]   (f32)
// Hoists the input projection out of the sequential recurrence.
// ---------------------------------------------------------------------------
__global__ void esn_uproj(const float* __restrict__ inp,
                          const float* __restrict__ Win,
                          const float* __restrict__ bias,
                          unsigned char* ws) {
  size_t idx = (size_t)blockIdx.x * blockDim.x + threadIdx.x;  // L*H threads
  int    j   = (int)(idx & (H - 1));
  size_t t   = idx >> 12;                                      // /H
  const float4* x = (const float4*)(inp + t * IN);
  const float4* w = (const float4*)(Win + (size_t)j * IN);
  float acc = bias[j];
#pragma unroll
  for (int q = 0; q < IN / 4; ++q) {
    float4 a = x[q], b = w[q];
    acc += a.x * b.x + a.y * b.y + a.z * b.z + a.w * b.w;
  }
  ((float*)(ws + WS_U))[idx] = acc;
}

// ---------------------------------------------------------------------------
// Persistent recurrence kernel. 32 WGs x 8 waves; wave w owns 16-row tile
// mt = wg*8 + w. Per step: 128 x v_wmma_f32_16x16x32_bf16 over K=4096 with a
// 2-deep register pipeline and 2 accumulators, tanh, bf16 writeback (double
// buffered on step parity), fused readout partial, grid-wide release/acquire
// barrier on a monotonic counter.
// ---------------------------------------------------------------------------
__global__ __launch_bounds__(TPB) void esn_recurrence(const float* __restrict__ w_ro,
                                                      unsigned char* ws) {
  const int wg   = blockIdx.x;
  const int wave = threadIdx.x >> 5;
  const int lane = threadIdx.x & 31;
  const int mt   = wg * 8 + wave;       // row-tile id, 0..255
  const int m0   = mt * 16;

  const v16bf* Asw      = (const v16bf*)(ws + WS_WH);
  __bf16*      hbuf     = (__bf16*)(ws + WS_H);
  const float* U        = (const float*)(ws + WS_U);
  float*       partials = (float*)(ws + WS_PART);
  unsigned int* counter = (unsigned int*)(ws + WS_COUNTER);

  __shared__ float lds_sum;

  // D layout (32-bit 16x16): lanes 0-15 hold rows m0..m0+7 in acc[0..7],
  // lanes 16-31 hold rows m0+8..m0+15. One writer lane per row group.
  const bool writer = (lane == 0) || (lane == 16);
  const int  rbase  = m0 + ((lane >= 16) ? 8 : 0);

  float wro[8];
#pragma unroll
  for (int r = 0; r < 8; ++r) wro[r] = w_ro[rbase + r];

  const v16bf* a_base = Asw + (size_t)mt * KT * 32 + lane;
  const int    hoff   = (lane >= 16) ? 16 : 0;   // B frag: lanes 0-15 K=0..15, 16-31 K=16..31

  for (int step = 0; step < L; ++step) {
    if (threadIdx.x == 0) lds_sum = 0.0f;
    __syncthreads();

    const __bf16* hread  = hbuf + (size_t)(step & 1) * H;
    __bf16*       hwrite = hbuf + (size_t)((step + 1) & 1) * H;

    // ---- software-pipelined WMMA loop (depth 2, dual accumulators) --------
    v8f acc0 = {}, acc1 = {};
    v16bf a0 = a_base[0];
    v16bf a1 = a_base[32];
    v16bf b0 = *(const v16bf*)(hread + 0 * 32 + hoff);
    v16bf b1 = *(const v16bf*)(hread + 1 * 32 + hoff);
#pragma unroll 2
    for (int kt = 0; kt < KT; kt += 2) {
      // branchless prefetch, wrapped so the address stays inside this tile
      __builtin_prefetch(
          (const void*)(a_base + (size_t)((kt + 16) & (KT - 1)) * 32), 0, 1);
      const int nk = (kt + 2) & (KT - 1);      // wrap: last iter reloads tile 0 (hot, unused)
      v16bf an0 = a_base[(size_t)nk * 32];
      v16bf an1 = a_base[(size_t)(nk + 1) * 32];
      v16bf bn0 = *(const v16bf*)(hread + nk * 32 + hoff);
      v16bf bn1 = *(const v16bf*)(hread + (nk + 1) * 32 + hoff);
      acc0 = __builtin_amdgcn_wmma_f32_16x16x32_bf16(
          false, a0, false, b0, (short)0, acc0, false, false);
      acc1 = __builtin_amdgcn_wmma_f32_16x16x32_bf16(
          false, a1, false, b1, (short)0, acc1, false, false);
      a0 = an0; a1 = an1; b0 = bn0; b1 = bn1;
    }

    // h_t = tanh(y + U_t); fused readout partial
    const float* u = U + (size_t)step * H + rbase;
    float hv[8];
#pragma unroll
    for (int r = 0; r < 8; ++r) hv[r] = tanhf(acc0[r] + acc1[r] + u[r]);

    if (writer) {
      float psum = 0.0f;
#pragma unroll
      for (int r = 0; r < 8; ++r) {
        hwrite[rbase + r] = (__bf16)hv[r];
        psum += hv[r] * wro[r];
      }
      atomicAdd(&lds_sum, psum);
    }
    __syncthreads();

    // grid-wide split barrier: release stores, signal, spin-acquire
    if (threadIdx.x == 0) {
      partials[(size_t)step * NWG + wg] = lds_sum;
      __threadfence();                 // release: h/partial stores visible device-wide
      atomicAdd(counter, 1u);
      const unsigned int target = (unsigned int)(step + 1) * (unsigned int)NWG;
      while (__hip_atomic_load(counter, __ATOMIC_ACQUIRE,
                               __HIP_MEMORY_SCOPE_AGENT) < target) {
        __builtin_amdgcn_s_sleep(2);
      }
    }
    __syncthreads();
    __threadfence();                   // acquire for all waves before reading h_t
  }
}

// ---------------------------------------------------------------------------
// Final: out[t] = b_ro + sum_wg partials[t][wg]
// ---------------------------------------------------------------------------
__global__ void esn_readout(const unsigned char* __restrict__ ws,
                            const float* __restrict__ b_ro,
                            float* __restrict__ out) {
  int t = blockIdx.x * blockDim.x + threadIdx.x;
  if (t >= L) return;
  const float* partials = (const float*)(ws + WS_PART);
  float s = b_ro[0];
#pragma unroll
  for (int w = 0; w < NWG; ++w) s += partials[(size_t)t * NWG + w];
  out[t] = s;
}

// ---------------------------------------------------------------------------
extern "C" void kernel_launch(void* const* d_in, const int* in_sizes, int n_in,
                              void* d_out, int out_size, void* d_ws, size_t ws_size,
                              hipStream_t stream) {
  const float* inp    = (const float*)d_in[0];  // (L,1,IN)
  const float* h_init = (const float*)d_in[1];  // (1,H)
  const float* W_in   = (const float*)d_in[2];  // (H,IN)
  const float* bias   = (const float*)d_in[3];  // (H)
  const float* W_h    = (const float*)d_in[4];  // (H,H)
  const float* w_ro   = (const float*)d_in[5];  // (H)
  const float* b_ro   = (const float*)d_in[6];  // scalar
  unsigned char* ws   = (unsigned char*)d_ws;
  float* out          = (float*)d_out;

  esn_setup<<<(2 * H + TPB - 1) / TPB, TPB, 0, stream>>>(h_init, ws);
  esn_swizzle_wh<<<(int)(((size_t)H * H) / TPB), TPB, 0, stream>>>(W_h, ws);
  esn_uproj<<<(int)(((size_t)L * H) / TPB), TPB, 0, stream>>>(inp, W_in, bias, ws);
  esn_recurrence<<<NWG, TPB, 0, stream>>>(w_ro, ws);
  esn_readout<<<(L + TPB - 1) / TPB, TPB, 0, stream>>>(ws, b_ro, out);
}
